// MultiHeadAttention_84250078478349
// MI455X (gfx1250) — compile-verified, matrix operand order checked by reading, not run on hardware
//
#include <hip/hip_runtime.h>
#include <hip/hip_bf16.h>

// MI455X / gfx1250, wave32. All contractions via v_wmma_f32_16x16x32_f16.

typedef __attribute__((ext_vector_type(16))) _Float16 v16h;
typedef __attribute__((ext_vector_type(8)))  float    v8f;

#define WMMA_F16(a,b,c) __builtin_amdgcn_wmma_f32_16x16x32_f16(false,(a),false,(b),(short)0,(c),false,false)

#define BB 2
#define HH 8
#define SS 2048
#define EE 512
#define DK 64
#define BH (BB*HH)
#define TT 257       // rel table rows
#define RST 272      // R row stride (padded)
#define WBW 288      // Wbin logical width (9 * 32)
#define PST 2056     // LDS sP row stride (halves), bank-decorrelated
#define WST 292      // LDS Wbin row stride (floats)
#define RVST 296     // LDS RVt row stride (halves)

// A/B fragment per ISA 7.12.2 (16-bit 16x32): lane m=l&15, hs=l>>4;
// elems 0..7 -> K = koff + 8*hs + e ; elems 8..15 -> K = koff + 16 + 8*hs + e
__device__ inline v16h frag_f16(const _Float16* row, int koff, int hs) {
  v16h r;
  const _Float16* p0 = row + koff + 8*hs;
  const _Float16* p1 = p0 + 16;
#pragma unroll
  for (int e = 0; e < 8; ++e) { r[e] = p0[e]; r[e+8] = p1[e]; }
  return r;
}
__device__ inline v16h frag_f32(const float* row, int koff, int hs) {
  v16h r;
  const float* p0 = row + koff + 8*hs;
  const float* p1 = p0 + 16;
#pragma unroll
  for (int e = 0; e < 8; ++e) { r[e] = (_Float16)p0[e]; r[e+8] = (_Float16)p1[e]; }
  return r;
}
__device__ inline v16h frag_zero() {
  v16h r;
#pragma unroll
  for (int e = 0; e < 16; ++e) r[e] = (_Float16)0.f;
  return r;
}

// ---------------- Kernel 1: fused QKV projection -------------------------
// grid (M/64, E/64, 3), block 256. Out: Qh/Kh [bh][s][d] f16, Vt [bh][d][s] f16.
__global__ void __launch_bounds__(256) k_qkv(
    const float* __restrict__ Xq, const float* __restrict__ Xk, const float* __restrict__ Xv,
    const float* __restrict__ Wq, const float* __restrict__ bq,
    const float* __restrict__ Wk, const float* __restrict__ bk,
    const float* __restrict__ Wv, const float* __restrict__ bv,
    _Float16* __restrict__ Qh, _Float16* __restrict__ Kh, _Float16* __restrict__ Vt)
{
  __shared__ _Float16 sA[64*40];   // A tile 64x32, stride 40 (bank-friendly)
  __shared__ _Float16 sB[64*40];   // W tile transposed: [n][k]
  const int tid = threadIdx.x, lane = tid & 31, wave = tid >> 5;
  const int hs = lane >> 4, ml = lane & 15;
  const int row0 = blockIdx.x * 64, n0 = blockIdx.y * 64, p = blockIdx.z;
  const float* X    = (p==0) ? Xq : (p==1) ? Xk : Xv;
  const float* W    = (p==0) ? Wq : (p==1) ? Wk : Wv;
  const float* bias = (p==0) ? bq : (p==1) ? bk : bv;

  const int msub = wave & 3;
  const int ns0  = (wave >> 2) * 2;
  v8f c0 = {}, c1 = {};

  for (int kk = 0; kk < EE; kk += 32) {
    { // stage A (f32 -> f16)
      int r = tid >> 2, cb = (tid & 3) * 8;
      const float* src = X + (size_t)(row0 + r) * EE + kk + cb;
      _Float16* dst = sA + r*40 + cb;
#pragma unroll
      for (int e = 0; e < 8; ++e) dst[e] = (_Float16)src[e];
    }
    { // stage W transposed
      int n = tid >> 2, kb = (tid & 3) * 8;
#pragma unroll
      for (int e = 0; e < 8; ++e)
        sB[n*40 + kb + e] = (_Float16)W[(size_t)(kk + kb + e) * EE + n0 + n];
    }
    __syncthreads();
    v16h a  = frag_f16(sA + (msub*16 + ml)*40, 0, hs);
    v16h b0 = frag_f16(sB + ((ns0+0)*16 + ml)*40, 0, hs);
    v16h b1 = frag_f16(sB + ((ns0+1)*16 + ml)*40, 0, hs);
    c0 = WMMA_F16(a, b0, c0);
    c1 = WMMA_F16(a, b1, c1);
    __syncthreads();
  }
#pragma unroll
  for (int t2 = 0; t2 < 2; ++t2) {
    v8f c = t2 ? c1 : c0;
    int nsub = ns0 + t2;
#pragma unroll
    for (int i = 0; i < 8; ++i) {
      int m = i + 8*hs;
      int r = row0 + msub*16 + m;             // global token row in [0, B*S)
      int n = n0 + nsub*16 + ml;              // global col in [0, 512)
      float v = c[i] + bias[n];
      int b = r / SS, s = r % SS, h = n >> 6, dd = n & 63;
      size_t bh = (size_t)(b*HH + h);
      _Float16 hv = (_Float16)v;
      if (p == 0)      Qh[(bh*SS + s)*DK + dd] = hv;
      else if (p == 1) Kh[(bh*SS + s)*DK + dd] = hv;
      else             Vt[(bh*DK + dd)*SS + s] = hv;   // V transposed
    }
  }
}

// ---------------- Kernel 2: R = Q @ rel_key_tab^T ------------------------
// one 16x16 tile per wave; grid.x = BH*(S/16)*17 / 8 waves, block 256.
__global__ void __launch_bounds__(256) k_relq(
    const _Float16* __restrict__ Qh, const float* __restrict__ relk,
    _Float16* __restrict__ R)
{
  const int tid = threadIdx.x, lane = tid & 31, wave = tid >> 5;
  const int hs = lane >> 4, ml = lane & 15;
  int wt = blockIdx.x * 8 + wave;
  int qsub = wt / 17, ttile = wt % 17;
  int bh = qsub / (SS/16), srow = (qsub % (SS/16)) * 16;
  int tcol = ttile*16 + ml;
  const _Float16* arow = Qh + ((size_t)bh*SS + srow + ml) * DK;
  v8f c = {};
#pragma unroll
  for (int ks = 0; ks < 2; ++ks) {
    v16h a = frag_f16(arow, ks*32, hs);
    v16h b = (tcol < TT) ? frag_f32(relk + (size_t)tcol*DK, ks*32, hs) : frag_zero();
    c = WMMA_F16(a, b, c);
  }
#pragma unroll
  for (int i = 0; i < 8; ++i) {
    int m = i + 8*hs;
    R[((size_t)bh*SS + srow + m)*RST + ttile*16 + ml] = (_Float16)c[i];
  }
}

// ---------------- Kernel 3: attention core -------------------------------
// one block per (b,h, 32-row q tile). Dynamic LDS ~207 KB.
__global__ void __launch_bounds__(256) k_attn(
    const _Float16* __restrict__ Qh, const _Float16* __restrict__ Kh,
    const _Float16* __restrict__ Vt, const _Float16* __restrict__ R,
    const float* __restrict__ mask, const float* __restrict__ relv,
    _Float16* __restrict__ AO)
{
  extern __shared__ __align__(16) char smem[];
  _Float16* sP   = (_Float16*)smem;                 // 32 x PST halves
  _Float16* sRVt = sP + (size_t)32*PST;             // 64 x RVST halves (rel_val^T)
  float*    sWb  = (float*)(sRVt + (size_t)64*RVST);// 32 x WST floats
  float*    sSum = sWb + (size_t)32*WST;            // 32 row sums

  const int tid = threadIdx.x, lane = tid & 31, wave = tid >> 5;
  const int hs = lane >> 4, ml = lane & 15;
  const int blk = blockIdx.x;
  const int bh = blk >> 6;                          // SS/32 == 64 q-tiles
  const int qt = blk & 63;
  const int q0 = qt * 32;
  const int b  = bh >> 3;
  const int h  = bh & 7;
  const float inv = 0.125f;                         // 1/sqrt(64)

  // stage rel_val transposed into LDS (f16), zero padding cols
  for (int idx = tid; idx < 64*RVST; idx += 256) {
    int dd = idx / RVST, t = idx % RVST;
    float v = (t < TT) ? relv[(size_t)t*DK + dd] : 0.f;
    sRVt[idx] = (_Float16)v;
  }

  // per-wave Q fragments (2 q-subtiles x 2 k-steps over d=64)
  v16h aQ[2][2];
#pragma unroll
  for (int qs = 0; qs < 2; ++qs)
#pragma unroll
    for (int ks = 0; ks < 2; ++ks)
      aQ[qs][ks] = frag_f16(Qh + ((size_t)bh*SS + q0 + qs*16 + ml)*DK, ks*32, hs);

  // phase 2: scores = (Q K^T + gather(R)) * inv + mask*(-1e9)  -> sP (f16)
  for (int kt = wave; kt < SS/16; kt += 8) {
    int k0 = kt * 16;
    v8f c0 = {}, c1 = {};
#pragma unroll
    for (int ks = 0; ks < 2; ++ks) {
      v16h bf = frag_f16(Kh + ((size_t)bh*SS + k0 + ml)*DK, ks*32, hs);
      c0 = WMMA_F16(aQ[0][ks], bf, c0);
      c1 = WMMA_F16(aQ[1][ks], bf, c1);
    }
#pragma unroll
    for (int qs = 0; qs < 2; ++qs) {
      v8f c = qs ? c1 : c0;
#pragma unroll
      for (int i = 0; i < 8; ++i) {
        int m  = i + 8*hs;
        int ql = qs*16 + m;
        int qg = q0 + ql;
        int kg = k0 + ml;
        int dlt = kg - qg; dlt = dlt < -128 ? -128 : (dlt > 128 ? 128 : dlt);
        float bias = (float)R[((size_t)bh*SS + qg)*RST + (dlt + 128)];
        float mval = mask[((size_t)b*SS + qg)*SS + kg];
        float s = (c[i] + bias) * inv + mval * (-1e9f);
        s = fmaxf(s, -60000.f);                    // keep finite in f16
        sP[(size_t)ql*PST + kg] = (_Float16)s;
      }
    }
  }
  __syncthreads();

  // phase 3: row softmax (store unnormalized exp; 1/sum folded into epilogue)
  for (int rr = 0; rr < 4; ++rr) {
    int q = wave*4 + rr;
    _Float16* row = sP + (size_t)q*PST;
    float mx = -3.0e38f;
    for (int kk = lane; kk < SS; kk += 32) mx = fmaxf(mx, (float)row[kk]);
#pragma unroll
    for (int off = 16; off > 0; off >>= 1) mx = fmaxf(mx, __shfl_xor(mx, off, 32));
    float sum = 0.f;
    for (int kk = lane; kk < SS; kk += 32) {
      float e = __expf((float)row[kk] - mx);
      row[kk] = (_Float16)e;
      sum += e;
    }
#pragma unroll
    for (int off = 16; off > 0; off >>= 1) sum += __shfl_xor(sum, off, 32);
    if (lane == 0) sSum[q] = sum;
  }

  // phase 4: Wbin[q][t] — shifted row copy + prefix(bin0)/suffix(bin256) sums
  for (int rr = 0; rr < 4; ++rr) {
    int q  = wave*4 + rr;
    int qg = q0 + q;
    const _Float16* row = sP + (size_t)q*PST;
    for (int j = lane; j < WBW; j += 32) {
      float v = 0.f;
      if (j >= 1 && j <= 255) {
        int kk = qg + j - 128;
        if (kk >= 0 && kk < SS) v = (float)row[kk];
      }
      sWb[(size_t)q*WST + j] = v;
    }
    float s0 = 0.f;
    for (int kk = lane; kk <= qg - 128; kk += 32) s0 += (float)row[kk];
#pragma unroll
    for (int off = 16; off > 0; off >>= 1) s0 += __shfl_xor(s0, off, 32);
    if (lane == 0) sWb[(size_t)q*WST + 0] = s0;
    float s1 = 0.f;
    for (int kk = qg + 128 + lane; kk < SS; kk += 32) s1 += (float)row[kk];
#pragma unroll
    for (int off = 16; off > 0; off >>= 1) s1 += __shfl_xor(s1, off, 32);
    if (lane == 0) sWb[(size_t)q*WST + 256] = s1;
  }
  __syncthreads();

  // phase 5: out = P@V + Wbin@rel_val  (fused accumulator), scale by 1/sum
  const int qs = wave >> 2, ds = wave & 3;
  const int dd = ds*16 + ml;
  v8f c = {};
  for (int kt = 0; kt < SS/32; ++kt) {
    v16h a  = frag_f16(sP + (size_t)(qs*16 + ml)*PST, kt*32, hs);
    v16h bf = frag_f16(Vt + ((size_t)bh*DK + dd)*SS, kt*32, hs);
    c = WMMA_F16(a, bf, c);
  }
#pragma unroll
  for (int kt = 0; kt < WBW/32; ++kt) {
    v16h a  = frag_f32(sWb + (size_t)(qs*16 + ml)*WST, kt*32, hs);
    v16h bf = frag_f16(sRVt + (size_t)dd*RVST, kt*32, hs);
    c = WMMA_F16(a, bf, c);
  }
#pragma unroll
  for (int i = 0; i < 8; ++i) {
    int m  = i + 8*hs;
    int ql = qs*16 + m;
    int qg = q0 + ql;
    float v = c[i] * (1.0f / sSum[ql]);
    AO[((size_t)b*SS + qg)*EE + h*64 + ds*16 + ml] = (_Float16)v;
  }
}

// ---------------- Kernel 4: output projection ----------------------------
__global__ void __launch_bounds__(256) k_out(
    const _Float16* __restrict__ AO, const float* __restrict__ Wo,
    const float* __restrict__ bo, float* __restrict__ out)
{
  __shared__ _Float16 sB[64*40];
  const int tid = threadIdx.x, lane = tid & 31, wave = tid >> 5;
  const int hs = lane >> 4, ml = lane & 15;
  const int row0 = blockIdx.x * 64, n0 = blockIdx.y * 64;
  const int msub = wave & 3, ns0 = (wave >> 2) * 2;
  v8f c0 = {}, c1 = {};
  const _Float16* arow = AO + (size_t)(row0 + msub*16 + ml) * EE;

  for (int kk = 0; kk < EE; kk += 32) {
    int n = tid >> 2, kb = (tid & 3) * 8;
#pragma unroll
    for (int e = 0; e < 8; ++e)
      sB[n*40 + kb + e] = (_Float16)Wo[(size_t)(kk + kb + e)*EE + n0 + n];
    __syncthreads();
    v16h a  = frag_f16(arow, kk, hs);
    v16h b0 = frag_f16(sB + ((ns0+0)*16 + ml)*40, 0, hs);
    v16h b1 = frag_f16(sB + ((ns0+1)*16 + ml)*40, 0, hs);
    c0 = WMMA_F16(a, b0, c0);
    c1 = WMMA_F16(a, b1, c1);
    __syncthreads();
  }
#pragma unroll
  for (int t2 = 0; t2 < 2; ++t2) {
    v8f c = t2 ? c1 : c0;
    int nsub = ns0 + t2;
#pragma unroll
    for (int i = 0; i < 8; ++i) {
      int m = i + 8*hs;
      int r = row0 + msub*16 + m;
      int n = n0 + nsub*16 + ml;
      out[(size_t)r*EE + n] = c[i] + bo[n];
    }
  }
}

extern "C" void kernel_launch(void* const* d_in, const int* in_sizes, int n_in,
                              void* d_out, int out_size, void* d_ws, size_t ws_size,
                              hipStream_t stream)
{
  (void)in_sizes; (void)n_in; (void)out_size; (void)ws_size;
  const float* q    = (const float*)d_in[0];
  const float* k    = (const float*)d_in[1];
  const float* v    = (const float*)d_in[2];
  const float* mask = (const float*)d_in[3];
  const float* Wq   = (const float*)d_in[4];
  const float* bq   = (const float*)d_in[5];
  const float* Wk   = (const float*)d_in[6];
  const float* bk   = (const float*)d_in[7];
  const float* Wv   = (const float*)d_in[8];
  const float* bv   = (const float*)d_in[9];
  const float* Wo   = (const float*)d_in[10];
  const float* bo   = (const float*)d_in[11];
  const float* relk = (const float*)d_in[12];
  const float* relv = (const float*)d_in[13];
  float* out = (float*)d_out;

  _Float16* Qh = (_Float16*)d_ws;                    // [BH][S][64]   4 MB
  _Float16* Kh = Qh + (size_t)BH*SS*DK;              // [BH][S][64]   4 MB
  _Float16* Vt = Kh + (size_t)BH*SS*DK;              // [BH][64][S]   4 MB (transposed)
  _Float16* R  = Vt + (size_t)BH*SS*DK;              // [BH][S][272] 17 MB
  _Float16* AO = R  + (size_t)BH*SS*RST;             // [B][S][512]   4 MB

  k_qkv <<<dim3(BB*SS/64, EE/64, 3), 256, 0, stream>>>(q,k,v,Wq,bq,Wk,bk,Wv,bv,Qh,Kh,Vt);
  k_relq<<<dim3((BH*(SS/16)*17)/8),  256, 0, stream>>>(Qh, relk, R);
  size_t smem = (size_t)32*PST*2 + (size_t)64*RVST*2 + (size_t)32*WST*4 + 32*4;
  k_attn<<<dim3(BH*(SS/32)), 256, smem, stream>>>(Qh,Kh,Vt,R,mask,relv,AO);
  k_out <<<dim3(BB*SS/64, EE/64), 256, 0, stream>>>(AO, Wo, bo, out);
}
